// BinaryLinear_81106162417900
// MI455X (gfx1250) — compile-verified, matrix operand order checked by reading, not run on hardware
//
#include <hip/hip_runtime.h>
#include <hip/hip_bf16.h>

typedef __bf16 bf16_t;
typedef __attribute__((ext_vector_type(16))) __bf16 v16bf;
typedef __attribute__((ext_vector_type(8)))  __bf16 v8bf;
typedef __attribute__((ext_vector_type(8)))  float  v8f;

// ---------------------------------------------------------------------------
// Pass 1: binarize weights fp32 -> bf16 {-1, 0, +1}. 768x768 = 589824 elems.
// Exactly representable in bf16, so the B matrix is lossless.
// ---------------------------------------------------------------------------
__global__ void binarize_w_kernel(const float* __restrict__ w,
                                  bf16_t* __restrict__ wb, int n) {
    int i = blockIdx.x * blockDim.x + threadIdx.x;
    if (i < n) {
        float v = w[i];
        float s = (v > 0.0f) ? 1.0f : ((v < 0.0f) ? -1.0f : 0.0f);
        wb[i] = (bf16_t)s;
    }
}

// ---------------------------------------------------------------------------
// Pass 2: GEMM  Out[M,N] = X[M,K] * Wb[N,K]^T   via v_wmma_f32_16x16x32_bf16
//   M = 32768, N = 768, K = 768
//   Block tile: 128(M) x 128(N), K-chunk 64, 256 threads = 8 waves (wave32)
//   Wave tile:  32(M) x 64(N)  -> 2x4 accumulators of 16x16
// ---------------------------------------------------------------------------
__global__ __launch_bounds__(256)
void bgemm_wmma_kernel(const float* __restrict__ X,
                       const bf16_t* __restrict__ Wb,
                       float* __restrict__ Out, int M) {
    constexpr int K  = 768;
    constexpr int N  = 768;
    constexpr int BM = 128;
    constexpr int BN = 128;
    constexpr int BK = 64;
    constexpr int SK = 72;   // LDS row pitch (bf16): 144B, 16B-aligned, bank-skewed

    __shared__ bf16_t As[BM * SK];   // x tile     (bf16)
    __shared__ bf16_t Bs[BN * SK];   // sign(W) tile

    const int tid   = threadIdx.x;
    const int lane  = tid & 31;
    const int wave  = tid >> 5;
    const int waveM = wave & 3;   // 4 waves along M: 32 rows each
    const int waveN = wave >> 2;  // 2 waves along N: 64 cols each

    const int gm0 = blockIdx.y * BM;   // M tile base (256 tiles)
    const int gn0 = blockIdx.x * BN;   // N tile base (6 tiles, fast dim -> L2 reuse of X)

    v8f acc[2][4];
#pragma unroll
    for (int i = 0; i < 2; ++i)
#pragma unroll
        for (int j = 0; j < 4; ++j)
            acc[i][j] = (v8f){0.f, 0.f, 0.f, 0.f, 0.f, 0.f, 0.f, 0.f};

    // fragment lane decomposition (ISA 16-bit A/B layout, wave32)
    const int fr    = lane & 15;          // row (A) / col (B) within 16
    const int khalf = (lane >> 4) * 8;    // lanes 16-31 hold K+8 / K+24 groups

    // global-load decompositions
    const int xc4 = (tid & 15) * 4;       // x: 16 float4 per 64-wide row
    const int xr0 = tid >> 4;             // x: rows 0..15, 8 passes of 16
    const int wc8 = (tid & 7) * 8;        // w: 8 x v8bf per 64-wide row
    const int wr0 = tid >> 3;             // w: rows 0..31, 4 passes of 32

    for (int kc = 0; kc < K; kc += BK) {
        // prefetch next x K-chunk into cache (global_prefetch_b8)
        if (kc + BK < K)
            __builtin_prefetch(X + (size_t)(gm0 + xr0) * K + (kc + BK) + xc4, 0, 3);

        // ---- stage x tile: fp32 global -> bf16 LDS -----------------------
#pragma unroll
        for (int p = 0; p < 8; ++p) {
            const int r = xr0 + p * 16;
            const float4 v = *reinterpret_cast<const float4*>(
                X + (size_t)(gm0 + r) * K + kc + xc4);
            bf16_t* dst = &As[r * SK + xc4];
            dst[0] = (bf16_t)v.x;
            dst[1] = (bf16_t)v.y;
            dst[2] = (bf16_t)v.z;
            dst[3] = (bf16_t)v.w;
        }
        // ---- stage sign(W) tile: bf16 global -> bf16 LDS -----------------
#pragma unroll
        for (int p = 0; p < 4; ++p) {
            const int r = wr0 + p * 32;
            const v8bf v = *reinterpret_cast<const v8bf*>(
                Wb + (size_t)(gn0 + r) * K + kc + wc8);
            *reinterpret_cast<v8bf*>(&Bs[r * SK + wc8]) = v;
        }
        __syncthreads();

        // ---- compute: 2 k-steps of 32, 8 WMMAs each ----------------------
#pragma unroll
        for (int kk = 0; kk < BK; kk += 32) {
            v16bf afrag[2];
            v16bf bfrag[4];
#pragma unroll
            for (int i = 0; i < 2; ++i) {
                const bf16_t* src = &As[(waveM * 32 + i * 16 + fr) * SK + kk + khalf];
                const v8bf lo = *reinterpret_cast<const v8bf*>(src);        // K+0..7  (or +8..15)
                const v8bf hi = *reinterpret_cast<const v8bf*>(src + 16);   // K+16..23 (or +24..31)
#pragma unroll
                for (int e = 0; e < 8; ++e) { afrag[i][e] = lo[e]; afrag[i][e + 8] = hi[e]; }
            }
#pragma unroll
            for (int j = 0; j < 4; ++j) {
                const bf16_t* src = &Bs[(waveN * 64 + j * 16 + fr) * SK + kk + khalf];
                const v8bf lo = *reinterpret_cast<const v8bf*>(src);
                const v8bf hi = *reinterpret_cast<const v8bf*>(src + 16);
#pragma unroll
                for (int e = 0; e < 8; ++e) { bfrag[j][e] = lo[e]; bfrag[j][e + 8] = hi[e]; }
            }
#pragma unroll
            for (int i = 0; i < 2; ++i)
#pragma unroll
                for (int j = 0; j < 4; ++j)
                    acc[i][j] = __builtin_amdgcn_wmma_f32_16x16x32_bf16(
                        /*neg_a=*/false, afrag[i],
                        /*neg_b=*/false, bfrag[j],
                        /*c_mod=*/(short)0, acc[i][j],
                        /*reuse_a=*/false, /*reuse_b=*/false);
        }
        __syncthreads();
    }

    // ---- epilogue: C/D layout -> global fp32 -----------------------------
    // VGPR v, lane<16 : (M = v,   N = lane);  lane>=16 : (M = v+8, N = lane-16)
    const int cg = lane & 15;
    const int rg = (lane >> 4) * 8;
#pragma unroll
    for (int i = 0; i < 2; ++i) {
#pragma unroll
        for (int j = 0; j < 4; ++j) {
            float* o = Out + (size_t)(gm0 + waveM * 32 + i * 16 + rg) * N
                           + (gn0 + waveN * 64 + j * 16 + cg);
#pragma unroll
            for (int v = 0; v < 8; ++v)
                o[(size_t)v * N] = acc[i][j][v];
        }
    }
}

// ---------------------------------------------------------------------------
extern "C" void kernel_launch(void* const* d_in, const int* in_sizes, int n_in,
                              void* d_out, int out_size, void* d_ws, size_t ws_size,
                              hipStream_t stream) {
    const float* x = (const float*)d_in[0];   // (4, 8192, 768) fp32
    const float* w = (const float*)d_in[1];   // (768, 768) fp32
    float* out = (float*)d_out;               // (4, 8192, 768) fp32

    const int K = 768, N = 768;
    const int wElems = in_sizes[1];           // 768*768
    const int M = in_sizes[0] / K;            // 32768

    bf16_t* wb = (bf16_t*)d_ws;               // 1.125 MB scratch

    binarize_w_kernel<<<(wElems + 255) / 256, 256, 0, stream>>>(w, wb, wElems);

    dim3 grid(N / 128, M / 128);              // (6, 256): N fastest -> x L2 reuse
    bgemm_wmma_kernel<<<grid, 256, 0, stream>>>(x, wb, out, M);
}
